// SparseFFN_50422916055410
// MI455X (gfx1250) — matrix operand order repaired and checked by reading, not compiled, not run on hardware
//
#include <hip/hip_runtime.h>
#include <cstdint>
#include <cstddef>

// ---------------------------------------------------------------------------
// MoE sparse FFN for MI455X (gfx1250, wave32, WMMA + async LDS staging)
// B=4 S=2048 D=1024 F=4096 E=8 K=2, capacity C = 4096
// ---------------------------------------------------------------------------

typedef __attribute__((ext_vector_type(16))) __bf16 v16bf;
typedef __attribute__((ext_vector_type(8)))  float  v8f;

#define TOKS 8192
#define DD   1024
#define FF   4096
#define EE   8
#define CAP  4096

// GEMM tiling
#define BM   128
#define BN   64
#define KC   64
#define PADK 72   // LDS k-stride (bf16 elems): 144B rows -> 16B aligned, conflict-poor

#if defined(__has_builtin)
#if __has_builtin(__builtin_amdgcn_global_load_async_to_lds_b128)
#define HAVE_ASYNC_LDS 1
#endif
#endif

__device__ __forceinline__ unsigned short f2bf(float f) {
  unsigned u = __float_as_uint(f);
  u += 0x7FFFu + ((u >> 16) & 1u);          // round to nearest even
  return (unsigned short)(u >> 16);
}

union FragU { uint4 q[2]; v16bf v; };

#ifdef HAVE_ASYNC_LDS
typedef int v4i_vs __attribute__((vector_size(16)));
typedef __attribute__((address_space(1))) v4i_vs* as1_v4i;
typedef __attribute__((address_space(3))) v4i_vs* as3_v4i;
#endif

// 16B global -> LDS copy; async (ASYNCcnt) when the CDNA5 builtin exists
__device__ __forceinline__ void copy16_g2lds(const unsigned short* g, unsigned short* l) {
#ifdef HAVE_ASYNC_LDS
  __builtin_amdgcn_global_load_async_to_lds_b128(
      (as1_v4i)(uintptr_t)g, (as3_v4i)(uintptr_t)l, 0, 0);
#else
  *(uint4*)l = *(const uint4*)g;
#endif
}

__device__ __forceinline__ void async_fence() {
#ifdef HAVE_ASYNC_LDS
#if __has_builtin(__builtin_amdgcn_s_wait_asynccnt)
  __builtin_amdgcn_s_wait_asynccnt(0);
#else
  asm volatile("s_wait_asynccnt 0x0" ::: "memory");
#endif
#endif
}

// ---------------------------------------------------------------------------
// 0. zero the output region
// ---------------------------------------------------------------------------
__global__ void zero_kernel(float4* __restrict__ p, int n4) {
  int i = blockIdx.x * blockDim.x + threadIdx.x;
  for (; i < n4; i += gridDim.x * blockDim.x) p[i] = make_float4(0.f, 0.f, 0.f, 0.f);
}

// ---------------------------------------------------------------------------
// 1. router: logits = x @ Wr + br ; softmax ; top-2
// ---------------------------------------------------------------------------
__global__ __launch_bounds__(256) void router_kernel(
    const float* __restrict__ x, const float* __restrict__ Wr,
    const float* __restrict__ br, float* __restrict__ logits_out,
    int* __restrict__ topk_idx, float* __restrict__ topk_probs) {
  __shared__ float xs[32][128];
  __shared__ float lg[32][8];
  const int e  = threadIdx.x & 7;
  const int lt = threadIdx.x >> 3;
  const int t0 = blockIdx.x * 32;

  float acc = 0.f;
  for (int c = 0; c < DD; c += 128) {
    for (int i = threadIdx.x; i < 32 * 128; i += 256) {
      int r = i >> 7, cc = i & 127;
      xs[r][cc] = x[(size_t)(t0 + r) * DD + c + cc];
    }
    __syncthreads();
    #pragma unroll 8
    for (int d = 0; d < 128; ++d)
      acc += xs[lt][d] * Wr[(size_t)(c + d) * EE + e];
    __syncthreads();
  }
  acc += br[e];
  lg[lt][e] = acc;
  const int t = t0 + lt;
  logits_out[(size_t)t * EE + e] = acc;
  __syncthreads();

  if (e == 0) {
    float m = lg[lt][0];
    for (int i = 1; i < EE; ++i) m = fmaxf(m, lg[lt][i]);
    float p[EE], s = 0.f;
    for (int i = 0; i < EE; ++i) { p[i] = __expf(lg[lt][i] - m); s += p[i]; }
    float inv = 1.f / s;
    int i0 = 0;
    for (int i = 1; i < EE; ++i) if (lg[lt][i] > lg[lt][i0]) i0 = i;
    int i1 = (i0 == 0) ? 1 : 0;
    for (int i = 0; i < EE; ++i) { if (i == i0) continue; if (lg[lt][i] > lg[lt][i1]) i1 = i; }
    topk_idx[t * 2 + 0]   = i0;
    topk_idx[t * 2 + 1]   = i1;
    topk_probs[t * 2 + 0] = p[i0] * inv;
    topk_probs[t * 2 + 1] = p[i1] * inv;
  }
}

// ---------------------------------------------------------------------------
// 2. per-expert top-capacity dispatch: bitonic sort in 64KB LDS
// ---------------------------------------------------------------------------
__global__ __launch_bounds__(1024) void dispatch_kernel(
    const float* __restrict__ logits, const int* __restrict__ topk_idx,
    const float* __restrict__ topk_probs, int* __restrict__ tok,
    float* __restrict__ w) {
  __shared__ unsigned long long keys[TOKS];   // 64 KB
  const int e = blockIdx.x;

  for (int t = threadIdx.x; t < TOKS; t += 1024) {
    int i0 = topk_idx[t * 2 + 0], i1 = topk_idx[t * 2 + 1];
    unsigned long long key = 0ull;
    if (i0 == e || i1 == e) {
      unsigned u = __float_as_uint(logits[(size_t)t * EE + e]);
      u = (u & 0x80000000u) ? ~u : (u | 0x80000000u);   // orderable
      key = ((unsigned long long)u << 16) | (unsigned)(TOKS - 1 - t);
    }
    keys[t] = key;
  }
  __syncthreads();

  for (int k = 2; k <= TOKS; k <<= 1) {
    for (int j = k >> 1; j > 0; j >>= 1) {
      for (int idx = threadIdx.x; idx < TOKS; idx += 1024) {
        int ixj = idx ^ j;
        if (ixj > idx) {
          bool desc = ((idx & k) == 0);
          unsigned long long a = keys[idx], b = keys[ixj];
          if (desc ? (a < b) : (a > b)) { keys[idx] = b; keys[ixj] = a; }
        }
      }
      __syncthreads();
    }
  }

  for (int c = threadIdx.x; c < CAP; c += 1024) {
    unsigned long long key = keys[c];
    int t = (TOKS - 1) - (int)(key & 0xFFFFull);
    float wv = 0.f;
    if (key != 0ull) {
      int i0 = topk_idx[t * 2 + 0], i1 = topk_idx[t * 2 + 1];
      wv = (i0 == e ? topk_probs[t * 2 + 0] : 0.f) +
           (i1 == e ? topk_probs[t * 2 + 1] : 0.f);
    } else {
      t = 0;  // weight is 0, contribution vanishes
    }
    tok[e * CAP + c] = t;
    w[e * CAP + c]   = wv;
  }
}

// ---------------------------------------------------------------------------
// 3. gather tokens into contiguous bf16 activation buffer Xe[E][C][D]
// ---------------------------------------------------------------------------
__global__ __launch_bounds__(256) void gather_kernel(
    const float* __restrict__ x, const int* __restrict__ tok,
    unsigned int* __restrict__ Xe) {
  const int row = blockIdx.x;               // e*CAP + c
  const int t   = tok[row];
  const float2* src = (const float2*)(x + (size_t)t * DD);
  unsigned int* dst = Xe + (size_t)row * (DD / 2);
  for (int i = threadIdx.x; i < DD / 2; i += 256) {
    float2 f = src[i];
    dst[i] = (unsigned)f2bf(f.x) | ((unsigned)f2bf(f.y) << 16);
  }
}

// ---------------------------------------------------------------------------
// 4. GEMM1: G = Xe@Wg+bg, V = Xe@Wv+bv, H = V*silu(G)  -> bf16 H[E][C][F]
//    block tile 128x64, 8 waves (4x2), wave tile 32x32, K-chunk 64
//    A tile: async global->LDS (ASYNCcnt); B tiles: fp32->bf16 transpose
// ---------------------------------------------------------------------------
__global__ __launch_bounds__(256) void gemm1_kernel(
    const unsigned short* __restrict__ Xe, const float* __restrict__ Wg,
    const float* __restrict__ bg, const float* __restrict__ Wv,
    const float* __restrict__ bv, unsigned short* __restrict__ H) {
  __shared__ unsigned short As[BM * PADK];      // 18 KB
  __shared__ unsigned short Bg[BN * PADK];      // 9 KB
  __shared__ unsigned short Bv[BN * PADK];      // 9 KB

  const int e  = blockIdx.z;
  const int m0 = blockIdx.y * BM;
  const int n0 = blockIdx.x * BN;
  const unsigned short* A = Xe + (size_t)e * CAP * DD;
  const float* Wge = Wg + (size_t)e * DD * FF;
  const float* Wve = Wv + (size_t)e * DD * FF;

  const int tid  = threadIdx.x;
  const int lane = tid & 31;
  const int wave = tid >> 5;
  const int wm   = (wave & 3) * 32;
  const int wn   = (wave >> 2) * 32;
  const int r    = lane & 15;
  const int hi   = lane >> 4;

  v8f accG[2][2] = {};
  v8f accV[2][2] = {};

  for (int k0 = 0; k0 < DD; k0 += KC) {
    // A tile 128xKC: direct global -> LDS (async when available)
    #pragma unroll
    for (int i = 0; i < 4; ++i) {
      int fd = (i * 256 + tid) * 4;              // flat dword in 128x32-dword tile
      int ar = fd >> 5, ac = fd & 31;            // ac: dword within row
      copy16_g2lds(A + (size_t)(m0 + ar) * DD + k0 + ac * 2, &As[ar * PADK + ac * 2]);
    }
    // prefetch next chunk of weights into GL2 while we convert this one
    if (k0 + KC < DD) {
      const float* pfg = &Wge[(size_t)(k0 + KC + (tid >> 2)) * FF + n0 + (tid & 3) * 16];
      const float* pfv = &Wve[(size_t)(k0 + KC + (tid >> 2)) * FF + n0 + (tid & 3) * 16];
      __builtin_prefetch(pfg, 0, 3);
      __builtin_prefetch(pfv, 0, 3);
    }
    // B tiles: fp32 -> bf16, transposed to [n][k]
    #pragma unroll
    for (int i = 0; i < 16; ++i) {
      int fl = i * 256 + tid;
      int kk = fl >> 6, nn = fl & 63;
      size_t gi = (size_t)(k0 + kk) * FF + n0 + nn;
      Bg[nn * PADK + kk] = f2bf(Wge[gi]);
      Bv[nn * PADK + kk] = f2bf(Wve[gi]);
    }
    async_fence();
    __syncthreads();

    #pragma unroll
    for (int ks = 0; ks < 2; ++ks) {
      const int kb = ks * 32;
      FragU aF[2], gF[2], vF[2];
      #pragma unroll
      for (int mi = 0; mi < 2; ++mi) {
        const unsigned short* p = &As[(wm + mi * 16 + r) * PADK + kb + hi * 8];
        aF[mi].q[0] = *(const uint4*)p;
        aF[mi].q[1] = *(const uint4*)(p + 16);
      }
      #pragma unroll
      for (int ni = 0; ni < 2; ++ni) {
        const unsigned short* pg = &Bg[(wn + ni * 16 + r) * PADK + kb + hi * 8];
        gF[ni].q[0] = *(const uint4*)pg;
        gF[ni].q[1] = *(const uint4*)(pg + 16);
        const unsigned short* pv = &Bv[(wn + ni * 16 + r) * PADK + kb + hi * 8];
        vF[ni].q[0] = *(const uint4*)pv;
        vF[ni].q[1] = *(const uint4*)(pv + 16);
      }
      #pragma unroll
      for (int mi = 0; mi < 2; ++mi)
        #pragma unroll
        for (int ni = 0; ni < 2; ++ni) {
          accG[mi][ni] = __builtin_amdgcn_wmma_f32_16x16x32_bf16(
              false, aF[mi].v, false, gF[ni].v, (short)0, accG[mi][ni], false, false);
          accV[mi][ni] = __builtin_amdgcn_wmma_f32_16x16x32_bf16(
              false, aF[mi].v, false, vF[ni].v, (short)0, accV[mi][ni], false, false);
        }
    }
    __syncthreads();
  }

  // epilogue: bias, silu, bf16 store
  #pragma unroll
  for (int mi = 0; mi < 2; ++mi)
    #pragma unroll
    for (int ni = 0; ni < 2; ++ni) {
      int n = n0 + wn + ni * 16 + r;
      float bgv = bg[(size_t)e * FF + n];
      float bvv = bv[(size_t)e * FF + n];
      #pragma unroll
      for (int v = 0; v < 8; ++v) {
        int m = m0 + wm + mi * 16 + hi * 8 + v;
        float g  = accG[mi][ni][v] + bgv;
        float vl = accV[mi][ni][v] + bvv;
        float h  = vl * (g / (1.f + __expf(-g)));
        H[((size_t)e * CAP + m) * FF + n] = f2bf(h);
      }
    }
}

// ---------------------------------------------------------------------------
// 5. GEMM2: out_e = (H@Wo + bo) * w ; scatter-add to output via f32 atomics
// ---------------------------------------------------------------------------
__global__ __launch_bounds__(256) void gemm2_kernel(
    const unsigned short* __restrict__ H, const float* __restrict__ Wo,
    const float* __restrict__ bo, const int* __restrict__ tok,
    const float* __restrict__ w, float* __restrict__ out) {
  __shared__ unsigned short As[BM * PADK];
  __shared__ unsigned short Bs[BN * PADK];

  const int e  = blockIdx.z;
  const int m0 = blockIdx.y * BM;
  const int n0 = blockIdx.x * BN;
  const unsigned short* A = H + (size_t)e * CAP * FF;
  const float* Woe = Wo + (size_t)e * FF * DD;

  const int tid  = threadIdx.x;
  const int lane = tid & 31;
  const int wave = tid >> 5;
  const int wm   = (wave & 3) * 32;
  const int wn   = (wave >> 2) * 32;
  const int r    = lane & 15;
  const int hi   = lane >> 4;

  v8f acc[2][2] = {};

  for (int k0 = 0; k0 < FF; k0 += KC) {
    #pragma unroll
    for (int i = 0; i < 4; ++i) {
      int fd = (i * 256 + tid) * 4;
      int ar = fd >> 5, ac = fd & 31;
      copy16_g2lds(A + (size_t)(m0 + ar) * FF + k0 + ac * 2, &As[ar * PADK + ac * 2]);
    }
    if (k0 + KC < FF) {
      __builtin_prefetch(&Woe[(size_t)(k0 + KC + (tid >> 2)) * DD + n0 + (tid & 3) * 16], 0, 3);
    }
    #pragma unroll
    for (int i = 0; i < 16; ++i) {
      int fl = i * 256 + tid;
      int kk = fl >> 6, nn = fl & 63;
      Bs[nn * PADK + kk] = f2bf(Woe[(size_t)(k0 + kk) * DD + n0 + nn]);
    }
    async_fence();
    __syncthreads();

    #pragma unroll
    for (int ks = 0; ks < 2; ++ks) {
      const int kb = ks * 32;
      FragU aF[2], bF[2];
      #pragma unroll
      for (int mi = 0; mi < 2; ++mi) {
        const unsigned short* p = &As[(wm + mi * 16 + r) * PADK + kb + hi * 8];
        aF[mi].q[0] = *(const uint4*)p;
        aF[mi].q[1] = *(const uint4*)(p + 16);
      }
      #pragma unroll
      for (int ni = 0; ni < 2; ++ni) {
        const unsigned short* pb = &Bs[(wn + ni * 16 + r) * PADK + kb + hi * 8];
        bF[ni].q[0] = *(const uint4*)pb;
        bF[ni].q[1] = *(const uint4*)(pb + 16);
      }
      #pragma unroll
      for (int mi = 0; mi < 2; ++mi)
        #pragma unroll
        for (int ni = 0; ni < 2; ++ni)
          acc[mi][ni] = __builtin_amdgcn_wmma_f32_16x16x32_bf16(
              false, aF[mi].v, false, bF[ni].v, (short)0, acc[mi][ni], false, false);
    }
    __syncthreads();
  }

  #pragma unroll
  for (int mi = 0; mi < 2; ++mi)
    #pragma unroll
    for (int ni = 0; ni < 2; ++ni) {
      int n = n0 + wn + ni * 16 + r;
      float bov = bo[(size_t)e * DD + n];
      #pragma unroll
      for (int v = 0; v < 8; ++v) {
        int m = m0 + wm + mi * 16 + hi * 8 + v;
        float val = (acc[mi][ni][v] + bov) * w[e * CAP + m];
        int   t   = tok[e * CAP + m];
        atomicAdd(&out[(size_t)t * DD + n], val);
      }
    }
}

// ---------------------------------------------------------------------------
// launch
// ---------------------------------------------------------------------------
extern "C" void kernel_launch(void* const* d_in, const int* in_sizes, int n_in,
                              void* d_out, int out_size, void* d_ws, size_t ws_size,
                              hipStream_t stream) {
  (void)in_sizes; (void)n_in; (void)out_size; (void)ws_size;
  const float* x  = (const float*)d_in[0];
  const float* Wr = (const float*)d_in[1];
  const float* br = (const float*)d_in[2];
  const float* Wg = (const float*)d_in[3];
  const float* bg = (const float*)d_in[4];
  const float* Wv = (const float*)d_in[5];
  const float* bv = (const float*)d_in[6];
  const float* Wo = (const float*)d_in[7];
  const float* bo = (const float*)d_in[8];

  // d_out = [output BxSxD fp32 | logits TxE fp32 | topk_idx TxK i32]
  float* out    = (float*)d_out;
  float* logits = out + (size_t)TOKS * DD;
  int*   topk   = (int*)(logits + (size_t)TOKS * EE);

  // workspace layout
  char* ws = (char*)d_ws;
  float*          probs = (float*)(ws);                             //  64 KB
  int*            tok   = (int*)(ws + 65536);                       // 128 KB
  float*          w     = (float*)(ws + 196608);                    // 128 KB
  unsigned short* Xe    = (unsigned short*)(ws + 327680);           //  64 MB
  unsigned short* Hbuf  = (unsigned short*)(ws + 327680 + (size_t)EE * CAP * DD * 2);  // 256 MB

  const int n4 = (TOKS * DD) / 4;
  zero_kernel<<<2048, 256, 0, stream>>>((float4*)out, n4);
  router_kernel<<<TOKS / 32, 256, 0, stream>>>(x, Wr, br, logits, topk, probs);
  dispatch_kernel<<<EE, 1024, 0, stream>>>(logits, topk, probs, tok, w);
  gather_kernel<<<EE * CAP, 256, 0, stream>>>(x, tok, (unsigned int*)Xe);
  gemm1_kernel<<<dim3(FF / BN, CAP / BM, EE), 256, 0, stream>>>(Xe, Wg, bg, Wv, bv, Hbuf);
  gemm2_kernel<<<dim3(DD / BN, CAP / BM, EE), 256, 0, stream>>>(Hbuf, Wo, bo, tok, w, out);
}
